// GCN_5085241278657
// MI455X (gfx1250) — compile-verified, matrix operand order checked by reading, not run on hardware
//
#include <hip/hip_runtime.h>
#include <hip/hip_bf16.h>
#include <stdint.h>

typedef __attribute__((ext_vector_type(16))) _Float16 v16h;
typedef __attribute__((ext_vector_type(8)))  float    v8f;

// ---------------------------------------------------------------------------
// Degree / normalization kernels
// ---------------------------------------------------------------------------
__global__ void gcn_init_deg(float* __restrict__ deg, int n) {
    int i = blockIdx.x * blockDim.x + threadIdx.x;
    if (i < n) deg[i] = 1.0f;                 // self-loop contributes 1
}

__global__ void gcn_count_deg(const int* __restrict__ dst, float* __restrict__ deg, int E) {
    int e = blockIdx.x * blockDim.x + threadIdx.x;
    if (e < E) atomicAdd(&deg[dst[e]], 1.0f);
}

__global__ void gcn_rsqrt_deg(float* __restrict__ deg, int n) {
    int i = blockIdx.x * blockDim.x + threadIdx.x;
    if (i < n) deg[i] = rsqrtf(deg[i]);       // deg >= 1 always (self loop)
}

// ---------------------------------------------------------------------------
// Pre-swizzle weight matrix B (K x N, f32 row-major) into wave32 WMMA
// fragment order as f16: for tile_n tn, k-step ks, lane l the 16 halves a
// lane feeds to v_wmma_f32_16x16x32_f16 are stored contiguously (32 bytes).
// Fragment K ordering (per ISA 16-bit A/B layout):
//   element i (v=i/2, s=i%2): k = (v<4 ? 2v : 16+2(v-4)) + s + (lane/16)*8
// ---------------------------------------------------------------------------
template<int K, int N>
__global__ void gcn_pack_B(const float* __restrict__ B, _Float16* __restrict__ Bp) {
    const int KS = K / 32, TN = N / 16;
    int t = blockIdx.x * blockDim.x + threadIdx.x;
    if (t >= TN * KS * 32) return;
    int lane = t & 31;
    int ks   = (t >> 5) % KS;
    int tn   = t / (32 * KS);
    int col  = tn * 16 + (lane & 15);
    int half = (lane >> 4) * 8;
    int k0   = ks * 32;
    _Float16* dstp = Bp + (size_t)t * 16;
#pragma unroll
    for (int v = 0; v < 8; ++v) {
        int kk = k0 + ((v < 4) ? 2 * v : 16 + 2 * (v - 4)) + half;
        dstp[2 * v]     = (_Float16)B[(size_t)kk * N + col];
        dstp[2 * v + 1] = (_Float16)B[(size_t)(kk + 1) * N + col];
    }
}

// ---------------------------------------------------------------------------
// Strip-mined WMMA GEMM: one wave computes a full 16-row x N output strip,
// so each A fragment (streamed from HBM) is loaded ONCE and reused across
// all N/16 B-tiles (8 independent back-to-back v_wmma per k-step).
//   BIAS_RELU : apply a = relu(A + bias[k]) on the fly (fuses prior layer's
//               bias+activation into the A-fragment load)
//   WRITE_SELF: also emit Cself = dinv[row]^2 * C (fused GCN self-loop init)
// ---------------------------------------------------------------------------
template<int K, int N, bool BIAS_RELU, bool WRITE_SELF>
__global__ void gcn_wmma_gemm(const float* __restrict__ A,
                              const _Float16* __restrict__ Bp,
                              const float* __restrict__ bias,
                              const float* __restrict__ dinv,
                              float* __restrict__ C,
                              float* __restrict__ Cself, int M) {
    const int KS = K / 32, TN = N / 16;
    int wave = (blockIdx.x * blockDim.x + threadIdx.x) >> 5;
    int lane = threadIdx.x & 31;
    int tiles_m = (M + 15) >> 4;
    if (wave >= tiles_m) return;                     // wave-uniform exit

    int rowA = wave * 16 + (lane & 15);
    if (rowA > M - 1) rowA = M - 1;                  // clamp (keeps EXEC full)
    int half = (lane >> 4) * 8;
    int col0 = lane & 15;

    v8f acc[TN];
#pragma unroll
    for (int tn = 0; tn < TN; ++tn) acc[tn] = (v8f){};

    const _Float16* bptr = Bp + (size_t)lane * 16;
    for (int ks = 0; ks < KS; ++ks) {
        v16h a;
#pragma unroll
        for (int v = 0; v < 8; ++v) {
            int kk = ks * 32 + ((v < 4) ? 2 * v : 16 + 2 * (v - 4)) + half;
            float2 af = *(const float2*)(A + (size_t)rowA * K + kk);
            if (BIAS_RELU) {
                af.x = fmaxf(af.x + bias[kk], 0.0f);
                af.y = fmaxf(af.y + bias[kk + 1], 0.0f);
            }
            a[2 * v]     = (_Float16)af.x;
            a[2 * v + 1] = (_Float16)af.y;
        }
#pragma unroll
        for (int tn = 0; tn < TN; ++tn) {
            v16h b = *(const v16h*)(bptr + ((size_t)(tn * KS + ks) * 32) * 16);
            acc[tn] = __builtin_amdgcn_wmma_f32_16x16x32_f16(
                /*neg_a=*/false, a, /*neg_b=*/false, b,
                /*c_mod=*/(short)0, acc[tn], /*reuse_a=*/false, /*reuse_b=*/false);
        }
    }
    int rbase = wave * 16 + (lane >> 4) * 8;
#pragma unroll
    for (int r = 0; r < 8; ++r) {
        int rr = rbase + r;
        if (rr < M) {
            float di2 = 0.0f;
            if (WRITE_SELF) { float di = dinv[rr]; di2 = di * di; }
#pragma unroll
            for (int tn = 0; tn < TN; ++tn) {
                float v = acc[tn][r];
                C[(size_t)rr * N + tn * 16 + col0] = v;
                if (WRITE_SELF) Cself[(size_t)rr * N + tn * 16 + col0] = di2 * v;
            }
        }
    }
}

// ---------------------------------------------------------------------------
// Self-loop init: agg[i][f] = dinv[i]^2 * h[i][f]  (layer-2 only; layer-1's
// is fused into GEMM1's epilogue)
// ---------------------------------------------------------------------------
template<int F>
__global__ void gcn_selfloop_init(const float* __restrict__ h,
                                  const float* __restrict__ dinv,
                                  float* __restrict__ agg, int n) {
    long long t = (long long)blockIdx.x * blockDim.x + threadIdx.x;
    if (t >= (long long)n * F) return;
    int i = (int)(t / F);
    float di = dinv[i];
    agg[t] = di * di * h[t];
}

// ---------------------------------------------------------------------------
// Edge scatter: one wave per edge; 32 lanes x (F/32) features.
// agg[dst] += dinv[src]*dinv[dst] * h[src]   (f32 atomics, L2-resident)
// ---------------------------------------------------------------------------
template<int F>
__global__ void gcn_edge_scatter(const int* __restrict__ src,
                                 const int* __restrict__ dst,
                                 const float* __restrict__ dinv,
                                 const float* __restrict__ h,
                                 float* __restrict__ agg, int E) {
    const int VEC = F / 32;
    int t    = blockIdx.x * blockDim.x + threadIdx.x;
    int e    = t >> 5;
    int lane = t & 31;
    if (e >= E) return;
    int s = src[e], d = dst[e];
    float w = dinv[s] * dinv[d];
    const float* hp = h   + (size_t)s * F + lane * VEC;
    float*       ap = agg + (size_t)d * F + lane * VEC;
    if (VEC == 4) {
        float4 v = *(const float4*)hp;
        atomicAdd(ap + 0, w * v.x);
        atomicAdd(ap + 1, w * v.y);
        atomicAdd(ap + 2, w * v.z);
        atomicAdd(ap + 3, w * v.w);
    } else {
        float2 v = *(const float2*)hp;
        atomicAdd(ap + 0, w * v.x);
        atomicAdd(ap + 1, w * v.y);
    }
}

// ---------------------------------------------------------------------------
// Head: logits = (h2agg + b2) @ Wo + bo (64 x 40), fused log_softmax per row.
// One thread per node; Wo/bo/b2 are L2-hot and wave-uniform (scalar-friendly).
// ---------------------------------------------------------------------------
__global__ void gcn_head_logsoftmax(const float* __restrict__ h,
                                    const float* __restrict__ b2,
                                    const float* __restrict__ Wo,
                                    const float* __restrict__ bo,
                                    float* __restrict__ out, int n) {
    const int HID = 64, OUT = 40;
    int i = blockIdx.x * blockDim.x + threadIdx.x;
    if (i >= n) return;
    float hrow[HID];
#pragma unroll
    for (int k = 0; k < HID; ++k) hrow[k] = h[(size_t)i * HID + k] + b2[k];
    float lg[OUT];
    float mx = -3.4e38f;
    for (int j = 0; j < OUT; ++j) {
        float acc = bo[j];
#pragma unroll
        for (int k = 0; k < HID; ++k) acc = fmaf(hrow[k], Wo[k * OUT + j], acc);
        lg[j] = acc;
        mx = fmaxf(mx, acc);
    }
    float se = 0.0f;
    for (int j = 0; j < OUT; ++j) se += __expf(lg[j] - mx);
    float lse = mx + __logf(se);
    for (int j = 0; j < OUT; ++j) out[(size_t)i * OUT + j] = lg[j] - lse;
}

// ---------------------------------------------------------------------------
// Launch: x, edge_index, W1, b1, W2, b2, Wo, bo
// ---------------------------------------------------------------------------
extern "C" void kernel_launch(void* const* d_in, const int* in_sizes, int n_in,
                              void* d_out, int out_size, void* d_ws, size_t ws_size,
                              hipStream_t stream) {
    const float* x  = (const float*)d_in[0];
    const int*   ei = (const int*)  d_in[1];
    const float* W1 = (const float*)d_in[2];
    const float* b1 = (const float*)d_in[3];
    const float* W2 = (const float*)d_in[4];
    const float* b2 = (const float*)d_in[5];
    const float* Wo = (const float*)d_in[6];
    const float* bo = (const float*)d_in[7];
    float* out = (float*)d_out;

    const int IN = 256, H1 = 128, H2 = 64;
    int N = in_sizes[0] / IN;
    int E = in_sizes[1] / 2;
    const int* src = ei;
    const int* dst = ei + E;

    // Workspace layout (reused across stages):
    //   deg/dinv : N floats
    //   buf1     : N*128 floats  (h1, then h2)
    //   buf2     : N*128 floats  (agg1, then agg2)
    //   Bp1,Bp2  : pre-swizzled f16 weights
    float* deg  = (float*)d_ws;
    float* buf1 = deg + N;
    float* buf2 = buf1 + (size_t)N * H1;
    uintptr_t p = (uintptr_t)(buf2 + (size_t)N * H1);
    p = (p + 63) & ~(uintptr_t)63;
    _Float16* Bp1 = (_Float16*)p;                    // 256*128 halves
    _Float16* Bp2 = Bp1 + (size_t)IN * H1;           // 128*64 halves

    const int BT = 256;
    int tiles_m = (N + 15) / 16;

    // 1) degree + dinv
    gcn_init_deg <<<(N + BT - 1) / BT, BT, 0, stream>>>(deg, N);
    gcn_count_deg<<<(E + BT - 1) / BT, BT, 0, stream>>>(dst, deg, E);
    gcn_rsqrt_deg<<<(N + BT - 1) / BT, BT, 0, stream>>>(deg, N);

    // 2) pack weights into WMMA fragment order (f16)
    {
        int t1 = (H1 / 16) * (IN / 32) * 32;
        gcn_pack_B<IN, H1><<<(t1 + BT - 1) / BT, BT, 0, stream>>>(W1, Bp1);
        int t2 = (H2 / 16) * (H1 / 32) * 32;
        gcn_pack_B<H1, H2><<<(t2 + BT - 1) / BT, BT, 0, stream>>>(W2, Bp2);
    }

    // 3) layer 1 GEMM (strip per wave): h1 = x @ W1 -> buf1,
    //    fused self-loop init: buf2 = dinv^2 * h1
    {
        long long thr = (long long)tiles_m * 32;
        gcn_wmma_gemm<IN, H1, false, true>
            <<<(unsigned)((thr + BT - 1) / BT), BT, 0, stream>>>(
                x, Bp1, nullptr, deg, buf1, buf2, N);
    }
    // 4) layer 1 edge scatter into buf2
    {
        long long ethr = (long long)E * 32;
        gcn_edge_scatter<H1><<<(unsigned)((ethr + BT - 1) / BT), BT, 0, stream>>>(
            src, dst, deg, buf1, buf2, E);
    }
    // 5) layer 2 GEMM: h2 = relu(agg1 + b1) @ W2 -> buf1 (bias+ReLU fused
    //    into the A-fragment load; each A element is read exactly once)
    {
        long long thr = (long long)tiles_m * 32;
        gcn_wmma_gemm<H1, H2, true, false>
            <<<(unsigned)((thr + BT - 1) / BT), BT, 0, stream>>>(
                buf2, Bp2, b1, nullptr, buf1, nullptr, N);
    }
    // 6) layer 2 self-loop init + edge scatter into buf2
    {
        long long thr = (long long)N * H2;
        gcn_selfloop_init<H2><<<(unsigned)((thr + BT - 1) / BT), BT, 0, stream>>>(
            buf1, deg, buf2, N);
        long long ethr = (long long)E * 32;
        gcn_edge_scatter<H2><<<(unsigned)((ethr + BT - 1) / BT), BT, 0, stream>>>(
            src, dst, deg, buf1, buf2, E);
    }
    // 7) head GEMM (+b2 fused) + log_softmax
    gcn_head_logsoftmax<<<(N + 127) / 128, 128, 0, stream>>>(buf2, b2, Wo, bo, out, N);
}
